// GRU_28793460752460
// MI455X (gfx1250) — matrix operand order repaired and checked by reading, not compiled, not run on hardware
//
#include <hip/hip_runtime.h>
#include <hip/hip_bf16.h>
#include <math.h>

typedef __bf16 bf16;
typedef __attribute__((ext_vector_type(16))) __bf16 v16bf;
typedef __attribute__((ext_vector_type(8)))  __bf16 v8bf;
typedef __attribute__((ext_vector_type(8)))  float  v8f;

#define T_LEN 2048
#define BATCH 64
#define HID   256
#define G3    768               // 3*H
#define M_ROWS (T_LEN*BATCH)    // 131072
#define HPAD  264               // padded LDS row stride for h (bf16), 16B aligned
#define KSPLIT 64               // K range of Wi/Wh staged in LDS (multiple of 32)
#define W_LP   72               // padded LDS row stride for staged weights (bf16)
#define NT     4                // j-tiles per wave

// fast saturating gate math: v_exp_f32 + v_rcp_f32 (ample for bf16 pipeline)
__device__ __forceinline__ float fsigmoid_(float x) {
  return __builtin_amdgcn_rcpf(1.0f + __expf(-x));
}
__device__ __forceinline__ float ftanh_(float x) {
  return 1.0f - 2.0f * __builtin_amdgcn_rcpf(1.0f + __expf(2.0f * x));
}

// ------------------------------------------------------------------ casts
__global__ void cast_f32_bf16(const float* __restrict__ s, bf16* __restrict__ d, int n) {
  int i = blockIdx.x * blockDim.x + threadIdx.x;
  if (i < n) d[i] = (bf16)s[i];
}

// ------------------------------------------------------------------ fused pipeline
// 3 persistent blocks (512 thr = 16 waves), one per GRU layer, each on its
// own WGP. Per step t: acc = x_t @ Wi^T + h @ Wh^T (r/z fused, n-gate halves
// separate), gates in registers, fp32 recurrent state, bf16 h_t published to
// the next layer with a release flag; consumer spins (acquire) AFTER its gh
// GEMM so flag latency hides behind the WMMA chain.
__global__ __launch_bounds__(512)
void gru_pipe(const bf16* __restrict__ x,     // [T][64][256] layer-0 input
              const bf16* __restrict__ Wib,   // [3][768][256]
              const bf16* __restrict__ Whb,   // [3][768][256]
              const float* __restrict__ bih,  // [3][768]
              const float* __restrict__ bhh,  // [3][768]
              bf16* __restrict__ out0,        // [T][64][256] layer-0 h
              bf16* __restrict__ out1,        // [T][64][256] layer-1 h
              float* __restrict__ dfinal,     // [64][256]
              int* __restrict__ flags)        // [2][T], pre-zeroed
{
  const int l = blockIdx.x;
  const bf16*  in   = (l == 0) ? x : (l == 1) ? out0 : out1;
  bf16*        outp = (l == 0) ? out0 : (l == 1) ? out1 : nullptr;
  const bf16*  Wi   = Wib + (size_t)l * G3 * HID;
  const bf16*  Wh   = Whb + (size_t)l * G3 * HID;
  const float* bi   = bih + (size_t)l * G3;
  const float* bh   = bhh + (size_t)l * G3;
  int* fin  = (l > 0) ? flags + (size_t)(l - 1) * T_LEN : nullptr;
  int* fout = (l < 2) ? flags + (size_t)l * T_LEN : nullptr;

  __shared__ bf16 hbuf[2][BATCH * HPAD];      // 2 x 33,792 B
  __shared__ bf16 wil[G3 * W_LP];             // 110,592 B: Wi[:,0:KSPLIT)
  __shared__ bf16 whl[G3 * W_LP];             // 110,592 B: Wh[:,0:KSPLIT)

  const int lane = threadIdx.x & 31;
  const int wave = threadIdx.x >> 5;          // 0..15
  const int ln16 = lane & 15;
  const int hi   = lane >> 4;
  const int m0   = (wave & 3) * 16;
  const int jb   = (wave >> 2) * NT;

  // one-time staging
  for (int i = threadIdx.x; i < G3 * (KSPLIT / 8); i += 512) {
    int r = i / (KSPLIT / 8);
    int c = (i % (KSPLIT / 8)) * 8;
    *(v8bf*)(wil + (size_t)r * W_LP + c) = *(const v8bf*)(Wi + (size_t)r * HID + c);
    *(v8bf*)(whl + (size_t)r * W_LP + c) = *(const v8bf*)(Wh + (size_t)r * HID + c);
  }
  for (int i = threadIdx.x; i < BATCH * HPAD; i += 512) hbuf[0][i] = (bf16)0.f;
  __syncthreads();

  // biases: r/z combined (bi+bh); n-gate halves separate
  float brz[2][NT], bin_[NT], bhn_[NT];
  #pragma unroll
  for (int jt = 0; jt < NT; ++jt) {
    int col = (jb + jt) * 16 + ln16;
    brz[0][jt] = bi[col] + bh[col];
    brz[1][jt] = bi[HID + col] + bh[HID + col];
    bin_[jt]   = bi[2 * HID + col];
    bhn_[jt]   = bh[2 * HID + col];
  }

  float hreg[NT][8];                          // fp32 recurrent state (own slots)
  #pragma unroll
  for (int jt = 0; jt < NT; ++jt)
    #pragma unroll
    for (int r = 0; r < 8; ++r) hreg[jt][r] = 0.f;

  v8f zero = {0.f,0.f,0.f,0.f,0.f,0.f,0.f,0.f};
  int p = 0;

  for (int t = 0; t < T_LEN; ++t) {
    v8f accRZ[2][NT], accIN[NT], accHN[NT];
    #pragma unroll
    for (int jt = 0; jt < NT; ++jt) {
      accRZ[0][jt] = zero; accRZ[1][jt] = zero; accIN[jt] = zero; accHN[jt] = zero;
    }

    // ---- pass A: gh = h @ Wh^T (no dependence on upstream flag)
    const bf16* hrow = &hbuf[p][(m0 + ln16) * HPAD];
    #pragma unroll
    for (int k0 = 0; k0 < HID; k0 += 32) {
      v8bf alo = *(const v8bf*)(hrow + k0 + hi * 16);          // ds_load_b128
      v8bf ahi = *(const v8bf*)(hrow + k0 + hi * 16 + 8);
      v16bf a = __builtin_shufflevector(alo, ahi, 0,1,2,3,4,5,6,7,8,9,10,11,12,13,14,15);
      #pragma unroll
      for (int jt = 0; jt < NT; ++jt) {
        #pragma unroll
        for (int g = 0; g < 3; ++g) {
          int wr = g * HID + (jb + jt) * 16 + ln16;
          v16bf b;
          if (k0 + 32 <= KSPLIT) {
            v8bf blo  = *(const v8bf*)(whl + (size_t)wr * W_LP + k0 + hi * 16);
            v8bf bhi2 = *(const v8bf*)(whl + (size_t)wr * W_LP + k0 + hi * 16 + 8);
            b = __builtin_shufflevector(blo, bhi2, 0,1,2,3,4,5,6,7,8,9,10,11,12,13,14,15);
          } else {
            b = *(const v16bf*)(Wh + (size_t)wr * HID + k0 + hi * 16);
          }
          v8f* dst = (g < 2) ? &accRZ[g][jt] : &accHN[jt];
          *dst = __builtin_amdgcn_wmma_f32_16x16x32_bf16(
              false, a, false, b, (short)0, *dst, false, false);
        }
      }
    }

    // ---- wait for upstream h_t (hidden behind pass A)
    if (l > 0) {
      while (__hip_atomic_load(&fin[t], __ATOMIC_ACQUIRE, __HIP_MEMORY_SCOPE_AGENT) == 0)
        __builtin_amdgcn_s_sleep(2);
    }

    // ---- pass B: gi = x_t @ Wi^T
    const bf16* xrow = in + ((size_t)t * BATCH + m0 + ln16) * HID;
    #pragma unroll
    for (int k0 = 0; k0 < HID; k0 += 32) {
      v8bf alo = *(const v8bf*)(xrow + k0 + hi * 16);          // global_load_b128
      v8bf ahi = *(const v8bf*)(xrow + k0 + hi * 16 + 8);
      v16bf a = __builtin_shufflevector(alo, ahi, 0,1,2,3,4,5,6,7,8,9,10,11,12,13,14,15);
      #pragma unroll
      for (int jt = 0; jt < NT; ++jt) {
        #pragma unroll
        for (int g = 0; g < 3; ++g) {
          int wr = g * HID + (jb + jt) * 16 + ln16;
          v16bf b;
          if (k0 + 32 <= KSPLIT) {
            v8bf blo  = *(const v8bf*)(wil + (size_t)wr * W_LP + k0 + hi * 16);
            v8bf bhi2 = *(const v8bf*)(wil + (size_t)wr * W_LP + k0 + hi * 16 + 8);
            b = __builtin_shufflevector(blo, bhi2, 0,1,2,3,4,5,6,7,8,9,10,11,12,13,14,15);
          } else {
            b = *(const v16bf*)(Wi + (size_t)wr * HID + k0 + hi * 16);
          }
          v8f* dst = (g < 2) ? &accRZ[g][jt] : &accIN[jt];
          *dst = __builtin_amdgcn_wmma_f32_16x16x32_bf16(
              false, a, false, b, (short)0, *dst, false, false);
        }
      }
    }

    // ---- gates (fast native math, all in registers)
    bf16* orow = outp ? outp + (size_t)t * BATCH * HID : nullptr;
    #pragma unroll
    for (int jt = 0; jt < NT; ++jt) {
      int col = (jb + jt) * 16 + ln16;
      #pragma unroll
      for (int r = 0; r < 8; ++r) {
        int b = m0 + r + hi * 8;
        float rg = fsigmoid_(accRZ[0][jt][r] + brz[0][jt]);
        float zg = fsigmoid_(accRZ[1][jt][r] + brz[1][jt]);
        float ng = ftanh_(accIN[jt][r] + bin_[jt] + rg * (accHN[jt][r] + bhn_[jt]));
        float hn = ng + zg * (hreg[jt][r] - ng);   // (1-z)*n + z*h
        hreg[jt][r] = hn;
        hbuf[p ^ 1][b * HPAD + col] = (bf16)hn;
        if (orow) orow[(size_t)b * HID + col] = (bf16)hn;
        if (l == 2 && t == T_LEN - 1) dfinal[b * HID + col] = hn;
      }
    }

    // ---- publish h_t to downstream layer, advance
    if (l < 2) __threadfence();        // per-thread stores agent-visible
    __syncthreads();                   // + intra-block happens-before
    if (l < 2 && threadIdx.x == 0)
      __hip_atomic_store(&fout[t], 1, __ATOMIC_RELEASE, __HIP_MEMORY_SCOPE_AGENT);
    p ^= 1;
  }
}

// ------------------------------------------------------------------ launch
extern "C" void kernel_launch(void* const* d_in, const int* in_sizes, int n_in,
                              void* d_out, int out_size, void* d_ws, size_t ws_size,
                              hipStream_t stream)
{
  (void)in_sizes; (void)n_in; (void)out_size; (void)ws_size;
  const float* x   = (const float*)d_in[0];
  const float* Wih = (const float*)d_in[1];
  const float* Whh = (const float*)d_in[2];
  const float* bih = (const float*)d_in[3];
  const float* bhh = (const float*)d_in[4];

  // workspace layout (~195 MB)
  bf16* xin  = (bf16*)d_ws;                                // [M][256]
  bf16* out0 = xin  + (size_t)M_ROWS * HID;                // [M][256]
  bf16* out1 = out0 + (size_t)M_ROWS * HID;                // [M][256]
  bf16* Wib  = out1 + (size_t)M_ROWS * HID;                // [3][768][256]
  bf16* Whb  = Wib  + (size_t)3 * G3 * HID;                // [3][768][256]
  int*  flags = (int*)(Whb + (size_t)3 * G3 * HID);        // [2][T]

  hipMemsetAsync(flags, 0, (size_t)2 * T_LEN * sizeof(int), stream);

  int nx = M_ROWS * HID;
  cast_f32_bf16<<<(nx + 255) / 256, 256, 0, stream>>>(x, xin, nx);
  int nw = 3 * G3 * HID;
  cast_f32_bf16<<<(nw + 255) / 256, 256, 0, stream>>>(Wih, Wib, nw);
  cast_f32_bf16<<<(nw + 255) / 256, 256, 0, stream>>>(Whh, Whb, nw);

  gru_pipe<<<3, 512, 0, stream>>>(xin, Wib, Whb, bih, bhh,
                                  out0, out1, (float*)d_out, flags);
}
